// DissipativeRINN_79912161509965
// MI455X (gfx1250) — compile-verified, matrix operand order checked by reading, not run on hardware
//
#include <hip/hip_runtime.h>
#include <cstdint>

// Sizes fixed by the reference
#define SDIM   16
#define NDIM   128
#define INDIM  16
#define OUTDIM 8
#define TSTEPS 128
#define FITER  30
#define DTC    0.01f
// LDS row pitch (in 32-bit words) for one K-pair row of 16 columns.
// 20 (not 16) so rows r and r+4 land in different 16-bank groups -> no
// 2-way conflict between the two lane-halves of a B-fragment read.
#define LDSTRIDE 20

typedef __attribute__((ext_vector_type(16))) _Float16 v16h;
typedef __attribute__((ext_vector_type(2)))  __fp16   v2fp16;
typedef __attribute__((ext_vector_type(8)))  float    v8f;

// 16-bit A-matrix 16x32 layout (ISA 7.12.2): lane m holds row M=m (both
// halves), VGPR v of half h holds K = base(v) + 8*h + 2*(v&3), pairs (K,K+1).
__device__ __forceinline__ int kpair(int v, int half) {
  return ((v < 4) ? 0 : 16) + 8 * half + 2 * (v & 3);
}

// One-instruction f32 pair -> packed f16 word (v_cvt_pk_rtz_f16_f32).
__device__ __forceinline__ uint32_t pkf16(float a, float b) {
  union { v2fp16 h; uint32_t u; } cv;
  cv.h = __builtin_amdgcn_cvt_pkrtz(a, b);
  return cv.u;
}

// 8 hardware tanhs back-to-back: consecutive TRANS ops are mutually
// independent, so only the last needs the 1-op hazard gap (v_nop).
__device__ __forceinline__ v8f tanh8(v8f x) {
  float r0, r1, r2, r3, r4, r5, r6, r7;
  asm volatile(
      "v_tanh_f32 %0, %8\n\t"
      "v_tanh_f32 %1, %9\n\t"
      "v_tanh_f32 %2, %10\n\t"
      "v_tanh_f32 %3, %11\n\t"
      "v_tanh_f32 %4, %12\n\t"
      "v_tanh_f32 %5, %13\n\t"
      "v_tanh_f32 %6, %14\n\t"
      "v_tanh_f32 %7, %15\n\t"
      "v_nop"
      : "=&v"(r0), "=&v"(r1), "=&v"(r2), "=&v"(r3),
        "=&v"(r4), "=&v"(r5), "=&v"(r6), "=&v"(r7)
      : "v"(x[0]), "v"(x[1]), "v"(x[2]), "v"(x[3]),
        "v"(x[4]), "v"(x[5]), "v"(x[6]), "v"(x[7]));
  v8f r = {r0, r1, r2, r3, r4, r5, r6, r7};
  return r;
}

__device__ __forceinline__ v8f wmma_f16(v16h a, v16h b, v8f c) {
  return __builtin_amdgcn_wmma_f32_16x16x32_f16(false, a, false, b,
                                                (short)0, c, false, false);
}

// Load a 32x16 f16 B-fragment from packed LDS ([K/2][16] words, pitch LDSTRIDE).
__device__ __forceinline__ v16h load_bfrag(const uint32_t* base, int col, int half) {
  union { v16h v; uint32_t u[8]; } r;
#pragma unroll
  for (int vv = 0; vv < 8; ++vv) {
    const int k = kpair(vv, half);                 // even K of the pair
    r.u[vv] = base[(k >> 1) * LDSTRIDE + col];
  }
  return r.v;
}

// Store a 16x16 f32 D-tile (lane<16: M=r, lane>=16: M=8+r, N=col) into packed
// f16 LDS at pair-row base `wordrowbase`.
__device__ __forceinline__ void store_dtile(uint32_t* base, const v8f& d,
                                            int col, int half, int wordrowbase) {
#pragma unroll
  for (int j = 0; j < 4; ++j)
    base[(wordrowbase + 4 * half + j) * LDSTRIDE + col] = pkf16(d[2 * j], d[2 * j + 1]);
}

__global__ __launch_bounds__(256) void rinn_fused(
    const float* __restrict__ obs,  const float* __restrict__ state0,
    const float* __restrict__ Am,   const float* __restrict__ Bw,
    const float* __restrict__ By,   const float* __restrict__ Cv,
    const float* __restrict__ Dvw,  const float* __restrict__ Dvy,
    const float* __restrict__ Cu,   const float* __restrict__ Duw,
    const float* __restrict__ Duy,  const float* __restrict__ logs,
    float* __restrict__ out, int Tlen) {
  // xy_pk: stacked [xT; yT] (K=0..31) packed f16 pairs; w_pk: ping-pong wT (K=0..127)
  __shared__ uint32_t xy_pk[16 * LDSTRIDE];
  __shared__ uint32_t w_pk[2][64 * LDSTRIDE];

  const int tid  = threadIdx.x;
  const int wave = tid >> 5;
  const int lane = tid & 31;
  const int col  = lane & 15;   // batch column within tile / A-frag row M
  const int half = lane >> 4;
  const int b0   = blockIdx.x * 16;
  const int mrow = 16 * wave + col;   // this wave's N-dim row slice

  // ---- persistent weight A-fragments (loaded once, reused 30*128 times) ----
  v16h fCvDvy, fDvw[4];
#pragma unroll
  for (int e = 0; e < 16; ++e) {
    const int k = kpair(e >> 1, half) + (e & 1);
    fCvDvy[e] = (_Float16)((k < 16) ? Cv[mrow * 16 + k] : Dvy[mrow * 16 + (k - 16)]);
  }
#pragma unroll
  for (int c = 0; c < 4; ++c)
#pragma unroll
    for (int e = 0; e < 16; ++e) {
      const int k = 32 * c + kpair(e >> 1, half) + (e & 1);
      fDvw[c][e] = (_Float16)Dvw[mrow * 128 + k];
    }

  // wave 0: output head [Cu|Duy] (+Duw); wave 1: state head [A|By] (+Bw)
  v16h fHead    = {};
  v16h fWmat[4] = {{0}, {0}, {0}, {0}};
  v8f  xf       = {};               // wave 1: f32 state, D-tile layout
  float4 lsA = {}, lsB = {};
  if (wave == 0) {
    if (col < OUTDIM) {             // rows >= 8 stay zero (pad)
#pragma unroll
      for (int e = 0; e < 16; ++e) {
        const int k = kpair(e >> 1, half) + (e & 1);
        fHead[e] = (_Float16)((k < 16) ? Cu[col * 16 + k] : Duy[col * 16 + (k - 16)]);
      }
#pragma unroll
      for (int c = 0; c < 4; ++c)
#pragma unroll
        for (int e = 0; e < 16; ++e) {
          const int k = 32 * c + kpair(e >> 1, half) + (e & 1);
          fWmat[c][e] = (_Float16)Duw[col * 128 + k];
        }
    }
    lsA = make_float4(logs[0], logs[1], logs[2], logs[3]);
    lsB = make_float4(logs[4], logs[5], logs[6], logs[7]);
  } else if (wave == 1) {
#pragma unroll
    for (int e = 0; e < 16; ++e) {
      const int k = kpair(e >> 1, half) + (e & 1);
      fHead[e] = (_Float16)((k < 16) ? Am[col * 16 + k] : By[col * 16 + (k - 16)]);
    }
#pragma unroll
    for (int c = 0; c < 4; ++c)
#pragma unroll
      for (int e = 0; e < 16; ++e) {
        const int k = 32 * c + kpair(e >> 1, half) + (e & 1);
        fWmat[c][e] = (_Float16)Bw[col * 128 + k];
      }
#pragma unroll
    for (int r = 0; r < 8; ++r)
      xf[r] = state0[(size_t)(b0 + col) * SDIM + (8 * half + r)];
    store_dtile(xy_pk, xf, col, half, 0);   // xT rows K=0..15
  }

  const v8f vzero = {};

  for (int t = 0; t < Tlen; ++t) {
    // ---- cooperative transposed load of y_t into xy rows K=16..31 ----
    {
      const int i  = tid & 15;
      const int bc = tid >> 4;
      const size_t g = ((size_t)(b0 + bc) * Tlen + t) * INDIM + i;
      const float v = obs[g];
      if (t + 1 < Tlen) __builtin_prefetch(&obs[g + INDIM], 0, 1);
      const int k = 16 + i;
      _Float16* hp = reinterpret_cast<_Float16*>(xy_pk);
      hp[((k >> 1) * LDSTRIDE + bc) * 2 + (k & 1)] = (_Float16)v;
    }
    __syncthreads();   // xT (from prev step / init) + yT visible to all waves

    // ---- b_m = [Cv|Dvy]_m @ [xT;yT] : held in registers for all 30 iters ----
    const v16h bxy = load_bfrag(xy_pk, col, half);
    const v8f  bacc = wmma_f16(fCvDvy, bxy, vzero);

    // ---- DEQ fixed point: wT = tanh(bT + Dvw @ wT), 30 iters, ping-pong LDS --
    v8f cur = tanh8(bacc);                       // iter 0: w=0 -> tanh(b)
    store_dtile(&w_pk[0][0], cur, col, half, 8 * wave);
    __syncthreads();

    int rd = 0;
#pragma unroll 1
    for (int it = 1; it < FITER; ++it) {
      v8f acc = bacc;
#pragma unroll
      for (int c = 0; c < 4; ++c) {
        const v16h wb = load_bfrag(&w_pk[rd][c * 16 * LDSTRIDE], col, half);
        acc = wmma_f16(fDvw[c], wb, acc);
      }
      cur = tanh8(acc);
      rd ^= 1;
      store_dtile(&w_pk[rd][0], cur, col, half, 8 * wave);
      __syncthreads();
    }
    // rd == 1 (29 flips); next timestep's iter-0 writes buffer 0 -> no WAR.

    if (wave == 0) {
      // uT = [Cu|Duy] @ [xT;yT] + Duw @ wT  (rows 0..7 valid)
      v8f u = wmma_f16(fHead, bxy, vzero);
#pragma unroll
      for (int c = 0; c < 4; ++c)
        u = wmma_f16(fWmat[c], load_bfrag(&w_pk[rd][c * 16 * LDSTRIDE], col, half), u);
      float* op = out + ((size_t)(b0 + col) * Tlen + t) * (2 * OUTDIM);
      if (half == 0) {   // lanes 0..15 hold M=0..7 = all 8 action means
        *reinterpret_cast<float4*>(op)     = make_float4(u[0], u[1], u[2], u[3]);
        *reinterpret_cast<float4*>(op + 4) = make_float4(u[4], u[5], u[6], u[7]);
      } else {           // lanes 16..31: broadcast log_stds for the same column
        *reinterpret_cast<float4*>(op + 8)  = lsA;
        *reinterpret_cast<float4*>(op + 12) = lsB;
      }
    } else if (wave == 1) {
      // dxT = [A|By] @ [xT;yT] + Bw @ wT ; Euler update kept in f32 registers
      v8f dx = wmma_f16(fHead, bxy, vzero);
#pragma unroll
      for (int c = 0; c < 4; ++c)
        dx = wmma_f16(fWmat[c], load_bfrag(&w_pk[rd][c * 16 * LDSTRIDE], col, half), dx);
#pragma unroll
      for (int r = 0; r < 8; ++r) xf[r] += DTC * dx[r];
      store_dtile(xy_pk, xf, col, half, 0);   // new xT for next timestep
    }
    // No extra barrier needed: xy_pk is next read only after the t+1 barrier.
  }
}

extern "C" void kernel_launch(void* const* d_in, const int* in_sizes, int n_in,
                              void* d_out, int out_size, void* d_ws, size_t ws_size,
                              hipStream_t stream) {
  const float* obs  = (const float*)d_in[0];
  const float* st0  = (const float*)d_in[1];
  const float* Am   = (const float*)d_in[2];
  const float* Bw   = (const float*)d_in[3];
  const float* By   = (const float*)d_in[4];
  const float* Cv   = (const float*)d_in[5];
  const float* Dvw  = (const float*)d_in[6];
  const float* Dvy  = (const float*)d_in[7];
  const float* Cu   = (const float*)d_in[8];
  const float* Duw  = (const float*)d_in[9];
  const float* Duy  = (const float*)d_in[10];
  const float* logs = (const float*)d_in[11];
  float* out = (float*)d_out;
  (void)d_ws; (void)ws_size; (void)n_in; (void)out_size;

  const int Tlen = TSTEPS;
  const int B = in_sizes[0] / (Tlen * INDIM);   // 4096
  dim3 grid(B / 16), block(256);
  hipLaunchKernelGGL(rinn_fused, grid, block, 0, stream,
                     obs, st0, Am, Bw, By, Cv, Dvw, Dvy, Cu, Duw, Duy, logs,
                     out, Tlen);
}
